// EulerCEMinimal_49581102465599
// MI455X (gfx1250) — matrix operand order repaired and checked by reading, not compiled, not run on hardware
//
#include <hip/hip_runtime.h>
#include <hip/hip_bf16.h>
#include <math.h>

// ---------------- types / helpers ----------------
typedef __attribute__((ext_vector_type(16))) __bf16 v16bf;
typedef __attribute__((ext_vector_type(8)))  float  v8f;
typedef __attribute__((ext_vector_type(4)))  int    v4i_;

#define NB  2
#define NH  16
#define SEQ 2048
#define DH  64
#define DM  1024

#if defined(__gfx1250__) && __has_builtin(__builtin_amdgcn_global_load_async_to_lds_b128) && __has_builtin(__builtin_amdgcn_s_wait_asynccnt)
#define USE_ASYNC_LDS 1
#else
#define USE_ASYNC_LDS 0
#endif

#if USE_ASYNC_LDS
typedef __attribute__((address_space(1))) v4i_* gv4p;
typedef __attribute__((address_space(3))) v4i_* lv4p;
#endif

__device__ __forceinline__ unsigned short f2bf(float f) {
    unsigned u = __float_as_uint(f);
    unsigned r = u + 0x7FFFu + ((u >> 16) & 1u);   // round-to-nearest-even
    return (unsigned short)(r >> 16);
}

__device__ __forceinline__ v16bf frag_from(uint4 lo, uint4 hi) {
    union { uint4 u[2]; v16bf v; } x;
    x.u[0] = lo; x.u[1] = hi;
    return x.v;
}

__device__ __forceinline__ v8f wmma_bf16(v16bf a, v16bf b, v8f c) {
    return __builtin_amdgcn_wmma_f32_16x16x32_bf16(false, a, false, b, (short)0, c,
                                                   false, false);
}

// 16-byte global -> LDS copy (async path tracked by ASYNCcnt when available)
__device__ __forceinline__ void copy16_g2l(const unsigned short* g, unsigned short* l) {
#if USE_ASYNC_LDS
    __builtin_amdgcn_global_load_async_to_lds_b128((gv4p)g, (lv4p)l, 0, 0);
#else
    *(uint4*)l = *(const uint4*)g;
#endif
}
__device__ __forceinline__ void async_join() {
#if USE_ASYNC_LDS
    __builtin_amdgcn_s_wait_asynccnt(0);
#endif
}

// ---------------- f32 -> bf16 convert ----------------
__global__ void k_f32_to_bf16(const float* __restrict__ in,
                              unsigned short* __restrict__ out, int n) {
    int i = blockIdx.x * blockDim.x + threadIdx.x;
    if (i < n) out[i] = f2bf(in[i]);
}

// ---------------- tiled bf16 GEMM: C(f32) = A(bf16 MxK) @ B(bf16 KxNt) ----------------
// Block tile 128(M) x 256(N), 8 waves in 2x4 grid; wave tile 64x64 (16 WMMA accums).
// K-step 64 -> 32 WMMAs per stage; 8 fragment loads per 16 WMMAs (0.5 fragment loads/WMMA).
// Compile-time Nt/K so strides fold into immediate offsets.
template<int Nt, int K>
__global__ __launch_bounds__(256) void k_gemm_bf16(
    const unsigned short* __restrict__ A,
    const unsigned short* __restrict__ Bm,
    float* __restrict__ C)
{
    __shared__ unsigned short lA[128 * 64];   // [m][k]       16 KB
    __shared__ unsigned short lB[256 * 64];   // transposed [n][k]  32 KB
    const int tid  = threadIdx.x;
    const int w    = tid >> 5, lane = tid & 31, lr = lane & 15, half = lane >> 4;
    const int wr   = w >> 2, wc = w & 3;       // 2 x 4 wave grid
    const int m0   = blockIdx.y * 128, n0 = blockIdx.x * 256;

    v8f acc[4][4] = {};

    for (int k0 = 0; k0 < K; k0 += 64) {
        // stage A: 128x64 bf16 = 1024 x 16B, 4 async copies per thread
        #pragma unroll
        for (int j = 0; j < 4; ++j) {
            int lin = tid + j * 256;            // 0..1023
            int row = lin >> 3, qc = lin & 7;
            copy16_g2l(&A[(size_t)(m0 + row) * K + k0 + qc * 8], &lA[row * 64 + qc * 8]);
        }
        // stage B transposed: 512 (k-pair, 16-wide n chunk) units, 2 per thread
        #pragma unroll
        for (int j = 0; j < 2; ++j) {
            int u = tid + j * 256;
            int kp = u >> 4, kk = kp * 2, nch = (u & 15) * 16;
            const unsigned short* s0 = &Bm[(size_t)(k0 + kk)     * Nt + n0 + nch];
            const unsigned short* s1 = &Bm[(size_t)(k0 + kk + 1) * Nt + n0 + nch];
            union { uint4 u4[2]; unsigned short s[16]; } e0, e1;
            e0.u4[0] = *(const uint4*)&s0[0]; e0.u4[1] = *(const uint4*)&s0[8];
            e1.u4[0] = *(const uint4*)&s1[0]; e1.u4[1] = *(const uint4*)&s1[8];
            unsigned* lBu = (unsigned*)lB;
            #pragma unroll
            for (int i = 0; i < 16; ++i)
                lBu[(nch + i) * 32 + kp] = (unsigned)e0.s[i] | ((unsigned)e1.s[i] << 16);
        }
        async_join();
        __syncthreads();

        #pragma unroll
        for (int kc = 0; kc < 2; ++kc) {
            v16bf af[4], bfr[4];
            #pragma unroll
            for (int mi = 0; mi < 4; ++mi) {
                int row = wr * 64 + mi * 16 + lr;
                uint4 lo = *(const uint4*)&lA[row * 64 + kc * 32 + half * 16];
                uint4 hi = *(const uint4*)&lA[row * 64 + kc * 32 + half * 16 + 8];
                af[mi] = frag_from(lo, hi);
            }
            #pragma unroll
            for (int ni = 0; ni < 4; ++ni) {
                int col = wc * 64 + ni * 16 + lr;
                uint4 lo = *(const uint4*)&lB[col * 64 + kc * 32 + half * 16];
                uint4 hi = *(const uint4*)&lB[col * 64 + kc * 32 + half * 16 + 8];
                bfr[ni] = frag_from(lo, hi);
            }
            #pragma unroll
            for (int mi = 0; mi < 4; ++mi)
                #pragma unroll
                for (int ni = 0; ni < 4; ++ni)
                    acc[mi][ni] = wmma_bf16(af[mi], bfr[ni], acc[mi][ni]);
        }
        __syncthreads();
    }

    // epilogue: one base per 16x16 tile; r*Nt folds into immediate offsets
    #pragma unroll
    for (int mi = 0; mi < 4; ++mi)
        #pragma unroll
        for (int ni = 0; ni < 4; ++ni) {
            float* cp = C + (size_t)(m0 + wr * 64 + mi * 16 + half * 8) * Nt
                          + (n0 + wc * 64 + ni * 16 + lr);
            #pragma unroll
            for (int r = 0; r < 8; ++r)
                cp[(size_t)r * Nt] = acc[mi][ni][r];
        }
}

// ---------------- RoPE + split qkv into bf16 [B,H,N,DH] ----------------
__global__ void k_rope_split(const float* __restrict__ qkv,
                             unsigned short* __restrict__ qb,
                             unsigned short* __restrict__ kb,
                             unsigned short* __restrict__ vb)
{
    int tid = blockIdx.x * blockDim.x + threadIdx.x;    // NB*NH*SEQ*32 pairs
    int i  = tid & 31;                 // pair index within head dim
    int n  = (tid >> 5) & (SEQ - 1);
    int bh = tid >> 16;                // 32*2048 = 1<<16
    int h  = bh & (NH - 1);
    int b  = bh >> 4;

    const float* rowp = qkv + (size_t)(b * SEQ + n) * 3 * DM;
    int cq = h * DH + 2 * i;
    float q0 = rowp[cq],          q1 = rowp[cq + 1];
    float k0 = rowp[DM + cq],     k1 = rowp[DM + cq + 1];
    float v0 = rowp[2 * DM + cq], v1 = rowp[2 * DM + cq + 1];

    if ((h >> 1) & 1) {
        float invf = __powf(10000.0f, -(float)i / 32.0f);
        float th = (float)n * invf;
        float s, c;
        __sincosf(th, &s, &c);
        float r0 = q0 * c - q1 * s, r1 = q0 * s + q1 * c;
        q0 = r0; q1 = r1;
        r0 = k0 * c - k1 * s; r1 = k0 * s + k1 * c;
        k0 = r0; k1 = r1;
    }
    size_t dst = ((size_t)bh * SEQ + n) * DH + 2 * i;
    qb[dst] = f2bf(q0); qb[dst + 1] = f2bf(q1);
    kb[dst] = f2bf(k0); kb[dst + 1] = f2bf(k1);
    vb[dst] = f2bf(v0); vb[dst + 1] = f2bf(v1);
}

// ---------------- flash attention: one (b,h), 128 q-rows per block ----------------
__global__ __launch_bounds__(256) void k_flash_attn(
    const unsigned short* __restrict__ qb,
    const unsigned short* __restrict__ kb,
    const unsigned short* __restrict__ vb,
    const float* __restrict__ log_xi,
    unsigned short* __restrict__ attn)   // [B*N, DM] bf16
{
    __shared__ unsigned short lK[32 * 64];       // [k][d]
    __shared__ unsigned short lV[64 * 32];       // transposed [d][k]
    __shared__ unsigned short lP[8 * 16 * 32];   // per-wave 16x32 P scratch

    const int tid  = threadIdx.x;
    const int w    = tid >> 5, lane = tid & 31, lr = lane & 15, half = lane >> 4;
    const int bh   = blockIdx.y, h = bh & (NH - 1), b = bh >> 4;
    const int qblk = blockIdx.x * 128;
    const int qbase = qblk + w * 16;
    const float coef = ((h & 1) ? 1.0f : 0.0f) / __expf(log_xi[h]);

    const unsigned short* qp = qb + (size_t)bh * SEQ * DH;
    const unsigned short* kp = kb + (size_t)bh * SEQ * DH;
    const unsigned short* vp = vb + (size_t)bh * SEQ * DH;

    // Q fragments (two 16x32 chunks over dh=64), held in registers across key loop
    v16bf aQ[2];
    #pragma unroll
    for (int c = 0; c < 2; ++c) {
        int row = qbase + lr;
        int cb = c * 32 + half * 16;
        uint4 lo = *(const uint4*)&qp[(size_t)row * DH + cb];
        uint4 hi = *(const uint4*)&qp[(size_t)row * DH + cb + 8];
        aQ[c] = frag_from(lo, hi);
    }

    v8f O[4] = {};
    float mrow[8], lrow[8];
    #pragma unroll
    for (int r = 0; r < 8; ++r) { mrow[r] = -3.0e38f; lrow[r] = 0.0f; }

    const int nkb = blockIdx.x * 4 + 4;   // causal: key blocks of 32 up to q-block end
    for (int kbi = 0; kbi < nkb; ++kbi) {
        const int k0 = kbi * 32;
        // stage K tile (row-major, async copy)
        {
            int row = tid >> 3, col = (tid & 7) * 8;
            copy16_g2l(&kp[(size_t)(k0 + row) * DH + col], &lK[row * 64 + col]);
        }
        // stage V tile transposed lV[d][k]: k-pairs packed into b32 stores
        {
            int p = tid >> 4, d0 = (tid & 15) * 4;      // 16 k-pairs x 16 d-chunks(4)
            const unsigned short* s0 = &vp[(size_t)(k0 + 2 * p)     * DH + d0];
            const unsigned short* s1 = &vp[(size_t)(k0 + 2 * p + 1) * DH + d0];
            union { uint2 u; unsigned short s[4]; } e0, e1;
            e0.u = *(const uint2*)s0;
            e1.u = *(const uint2*)s1;
            unsigned* lVu = (unsigned*)lV;
            #pragma unroll
            for (int j = 0; j < 4; ++j)
                lVu[(d0 + j) * 16 + p] = (unsigned)e0.s[j] | ((unsigned)e1.s[j] << 16);
        }
        async_join();
        __syncthreads();

        // S = Q K^T : two 16x16 key tiles, each accumulated over dh via 2 WMMAs
        v8f sc[2];
        #pragma unroll
        for (int nt = 0; nt < 2; ++nt) {
            v8f s = {};
            #pragma unroll
            for (int c = 0; c < 2; ++c) {
                int keyl = nt * 16 + lr;
                int base = keyl * 64 + c * 32 + half * 16;
                uint4 lo = *(const uint4*)&lK[base];
                uint4 hi = *(const uint4*)&lK[base + 8];
                s = wmma_bf16(aQ[c], frag_from(lo, hi), s);
            }
            sc[nt] = s;
        }

        // scale + decay bias + causal mask + online softmax (row stats per C layout)
        #pragma unroll
        for (int r = 0; r < 8; ++r) {
            int qrow = qbase + r + half * 8;
            float sv0, sv1;
            {
                int key = k0 + lr;
                float x = sc[0][r] * 0.125f;
                sv0 = x + ((key <= qrow) ? (-(float)(qrow - key) * coef) : -3.0e38f);
                key = k0 + 16 + lr;
                x = sc[1][r] * 0.125f;
                sv1 = x + ((key <= qrow) ? (-(float)(qrow - key) * coef) : -3.0e38f);
            }
            float rm = fmaxf(sv0, sv1);
            #pragma unroll
            for (int m = 1; m < 16; m <<= 1) rm = fmaxf(rm, __shfl_xor(rm, m, 32));
            float mnew = fmaxf(mrow[r], rm);
            float corr = __expf(mrow[r] - mnew);
            float p0 = __expf(sv0 - mnew);
            float p1 = __expf(sv1 - mnew);
            float rs = p0 + p1;
            #pragma unroll
            for (int m = 1; m < 16; m <<= 1) rs += __shfl_xor(rs, m, 32);
            lrow[r] = lrow[r] * corr + rs;
            mrow[r] = mnew;
            #pragma unroll
            for (int dt = 0; dt < 4; ++dt) O[dt][r] *= corr;
            // C-layout -> LDS (bf16) for A-layout reload (same-wave, no barrier needed)
            int prow = (w * 16 + r + half * 8) * 32;
            lP[prow + lr]      = f2bf(p0);
            lP[prow + 16 + lr] = f2bf(p1);
        }

        // P (16x32) as A fragment, V tiles as B fragments: O += P V
        {
            int base = w * 512 + lr * 32 + half * 16;
            uint4 lo = *(const uint4*)&lP[base];
            uint4 hi = *(const uint4*)&lP[base + 8];
            v16bf aP = frag_from(lo, hi);
            #pragma unroll
            for (int dt = 0; dt < 4; ++dt) {
                int d = dt * 16 + lr;
                uint4 blo = *(const uint4*)&lV[d * 32 + half * 16];
                uint4 bhi = *(const uint4*)&lV[d * 32 + half * 16 + 8];
                O[dt] = wmma_bf16(aP, frag_from(blo, bhi), O[dt]);
            }
        }
        __syncthreads();
    }

    // normalize and write bf16 attn output; row stride DM is compile-time
    #pragma unroll
    for (int dt = 0; dt < 4; ++dt) {
        unsigned short* op = attn + (size_t)(b * SEQ + qbase + half * 8) * DM
                                  + h * DH + dt * 16 + lr;
        #pragma unroll
        for (int r = 0; r < 8; ++r)
            op[(size_t)r * DM] = f2bf(O[dt][r] / lrow[r]);
    }
}

// ---------------- launch ----------------
extern "C" void kernel_launch(void* const* d_in, const int* in_sizes, int n_in,
                              void* d_out, int out_size, void* d_ws, size_t ws_size,
                              hipStream_t stream)
{
    const float* x      = (const float*)d_in[0];
    const float* w_qkv  = (const float*)d_in[1];
    const float* w_o    = (const float*)d_in[2];
    const float* log_xi = (const float*)d_in[3];
    float* out = (float*)d_out;

    char* ws = (char*)d_ws;
    size_t off = 0;
    auto alloc = [&](size_t bytes) -> char* {
        char* p = ws + off;
        off += (bytes + 255) & ~(size_t)255;
        return p;
    };
    unsigned short* x_bf    = (unsigned short*)alloc((size_t)NB * SEQ * DM * 2);
    unsigned short* wqkv_bf = (unsigned short*)alloc((size_t)DM * 3 * DM * 2);
    unsigned short* wo_bf   = (unsigned short*)alloc((size_t)DM * DM * 2);
    float*          qkv_f   = (float*)alloc((size_t)NB * SEQ * 3 * DM * 4);
    unsigned short* q_bf    = (unsigned short*)alloc((size_t)NB * NH * SEQ * DH * 2);
    unsigned short* k_bf    = (unsigned short*)alloc((size_t)NB * NH * SEQ * DH * 2);
    unsigned short* v_bf    = (unsigned short*)alloc((size_t)NB * NH * SEQ * DH * 2);
    unsigned short* attn_bf = (unsigned short*)alloc((size_t)NB * SEQ * DM * 2);

    int nx = NB * SEQ * DM;
    k_f32_to_bf16<<<(nx + 255) / 256, 256, 0, stream>>>(x, x_bf, nx);
    int nwq = DM * 3 * DM;
    k_f32_to_bf16<<<(nwq + 255) / 256, 256, 0, stream>>>(w_qkv, wqkv_bf, nwq);
    int nwo = DM * DM;
    k_f32_to_bf16<<<(nwo + 255) / 256, 256, 0, stream>>>(w_o, wo_bf, nwo);

    dim3 g1(3 * DM / 256, NB * SEQ / 128);
    k_gemm_bf16<3 * DM, DM><<<g1, 256, 0, stream>>>(x_bf, wqkv_bf, qkv_f);

    int nrope = NB * NH * SEQ * 32;
    k_rope_split<<<nrope / 256, 256, 0, stream>>>(qkv_f, q_bf, k_bf, v_bf);

    dim3 g2(SEQ / 128, NB * NH);
    k_flash_attn<<<g2, 256, 0, stream>>>(q_bf, k_bf, v_bf, log_xi, attn_bf);

    dim3 g3(DM / 256, NB * SEQ / 128);
    k_gemm_bf16<DM, DM><<<g3, 256, 0, stream>>>(attn_bf, wo_bf, out);
}
